// SparseExplorerRouting_76441827934269
// MI455X (gfx1250) — matrix operand order repaired and checked by reading, not compiled, not run on hardware
//
#include <hip/hip_runtime.h>
#include <stdint.h>

// ---------------------------------------------------------------------------
// SparseExplorerRouting for MI455X (gfx1250, wave32, WMMA).
//
// Design (reasoned from MI455X specs, compile-only):
//  * 40,960 walk-steps, each needing a 33x1024 mat-vec against a 16 MB fp16
//    table -> entirely L2-resident (192 MB L2). L2-bandwidth bound, ~2.7 GB
//    total -> fp16 storage halves traffic vs fp32 and feeds the WMMA pipes.
//  * One wave32 per walk; sims computed with v_wmma_f32_16x16x32_f16:
//    A = window-row tiles (16x32 fp16 fragments, b128 loads), B = cur vector
//    replicated across columns, staged in LDS once per step (removes 16x
//    redundant global loads of the cur row). fp32 WMMA accumulation.
//  * Bit-faithful JAX Threefry-2x32 RNG chain (split / Gumbel categorical /
//    randint) so the sampled trajectories match the reference.
// ---------------------------------------------------------------------------

typedef __attribute__((ext_vector_type(16))) _Float16 v16h;
typedef __attribute__((ext_vector_type(8)))  _Float16 v8h;
typedef __attribute__((ext_vector_type(4)))  _Float16 v4h;
typedef __attribute__((ext_vector_type(8)))  float    v8f;

#define SEQ_LEN   8192
#define HIDDEN    1024
#define NUM_VIOL  1024
#define NUM_WALKS 5
#define WALK_LEN  8
#define NWALK_TOT (NUM_VIOL * NUM_WALKS)
#define HALF_WIN  16
#define WIN       33
#define TEMPF     5.0f
#define NEGL      (-1e9f)
#define EPS8      1e-8f

__device__ __forceinline__ int clampi(int x, int lo, int hi) {
    return x < lo ? lo : (x > hi ? hi : x);
}

// ----------------------- Threefry-2x32 (JAX-compatible) --------------------
__device__ __forceinline__ uint32_t rotl32(uint32_t v, uint32_t d) {
    return (v << d) | (v >> (32u - d));
}

__device__ __forceinline__ void tf2x32(uint32_t k0, uint32_t k1,
                                       uint32_t x0, uint32_t x1,
                                       uint32_t& o0, uint32_t& o1) {
    const uint32_t k2 = k0 ^ k1 ^ 0x1BD11BDAu;
    x0 += k0; x1 += k1;
#define TF_RND(r) { x0 += x1; x1 = rotl32(x1, r); x1 ^= x0; }
    TF_RND(13) TF_RND(15) TF_RND(26) TF_RND(6)  x0 += k1; x1 += k2 + 1u;
    TF_RND(17) TF_RND(29) TF_RND(16) TF_RND(24) x0 += k2; x1 += k0 + 2u;
    TF_RND(13) TF_RND(15) TF_RND(26) TF_RND(6)  x0 += k0; x1 += k1 + 3u;
    TF_RND(17) TF_RND(29) TF_RND(16) TF_RND(24) x0 += k1; x1 += k2 + 4u;
    TF_RND(13) TF_RND(15) TF_RND(26) TF_RND(6)  x0 += k2; x1 += k0 + 5u;
#undef TF_RND
    o0 = x0; o1 = x1;
}

// JAX gumbel: -log(-log(uniform(tiny, 1))), uniform from high 23 bits.
__device__ __forceinline__ float gumbel_from_bits(uint32_t bits) {
    const float TINY = 1.17549435e-38f;
    float f = __uint_as_float((bits >> 9) | 0x3f800000u) - 1.0f;  // [0,1)
    float u = fmaxf(TINY, f + TINY);   // (1-tiny) rounds to 1 in fp32
    return -logf(-logf(u));
}

// ----------------------------- preprocessing -------------------------------
// Per row: row_sq = |h|^2 (fp32) and normalized row in fp16.
__global__ __launch_bounds__(256)
void prep_norm_kernel(const float* __restrict__ hs,
                      _Float16* __restrict__ h16,
                      float* __restrict__ rsq) {
    __shared__ float red[256];
    __shared__ float s_inv;
    const int row = blockIdx.x;
    const int t   = threadIdx.x;
    const float4 x = ((const float4*)(hs + (size_t)row * HIDDEN))[t];
    red[t] = x.x * x.x + x.y * x.y + x.z * x.z + x.w * x.w;
    __syncthreads();
    for (int off = 128; off > 0; off >>= 1) {
        if (t < off) red[t] += red[t + off];
        __syncthreads();
    }
    if (t == 0) {
        rsq[row] = red[0];
        s_inv = 1.0f / (sqrtf(red[0]) + EPS8);
    }
    __syncthreads();
    const float inv = s_inv;
    v4h o;
    o[0] = (_Float16)(x.x * inv);
    o[1] = (_Float16)(x.y * inv);
    o[2] = (_Float16)(x.z * inv);
    o[3] = (_Float16)(x.w * inv);
    *(v4h*)(h16 + (size_t)row * HIDDEN + t * 4) = o;
}

__global__ void zero_out_kernel(float* __restrict__ out, int n) {
    int i = blockIdx.x * blockDim.x + threadIdx.x;
    if (i < n) out[i] = 0.0f;
}

// ------------------------------- walk kernel -------------------------------
__device__ __forceinline__ float logit_at(const float* s_sims, int cur, int p) {
    const int idx = cur - HALF_WIN + p;
    const bool valid = (idx >= 0) && (idx < SEQ_LEN) && (idx != cur);
    return valid ? (s_sims[p] * TEMPF) : (NEGL * TEMPF);
}

__global__ __launch_bounds__(32)
void walk_kernel(const _Float16* __restrict__ h16,
                 const float* __restrict__ rsq,
                 const int* __restrict__ viol,
                 float* __restrict__ out) {
    __shared__ _Float16 s_cur[HIDDEN] __attribute__((aligned(16)));
    __shared__ float s_sims[48];
    __shared__ float s_score[36];
    __shared__ int   s_pick;

    const int walk = blockIdx.x;            // 0..5119  (= v*5 + w)
    const int v    = walk / NUM_WALKS;
    const int lane = threadIdx.x;           // wave32
    const int m    = lane & 15;
    const int hi   = lane >> 4;             // which K-half this lane holds

    // --- derive this walk's key: jax.random.split(key(42), 5120)[walk] ---
    uint32_t key0, key1;
    {
        const uint32_t mt = NWALK_TOT;
        const uint32_t i  = (uint32_t)walk;
        uint32_t a0, a1, b0, b1;
        if (2u * i < mt) {                   // both bits from y0 half
            tf2x32(0u, 42u, 2u * i,       mt + 2u * i,      a0, a1);
            tf2x32(0u, 42u, 2u * i + 1u,  mt + 2u * i + 1u, b0, b1);
            key0 = a0; key1 = b0;
        } else {                             // both bits from y1 half
            tf2x32(0u, 42u, 2u * i - mt,      2u * i,       a0, a1);
            tf2x32(0u, 42u, 2u * i + 1u - mt, 2u * i + 1u,  b0, b1);
            key0 = a1; key1 = b1;
        }
    }

    const int start = viol[v];
    int   cur = start, prev = start, plen = 1;
    float min_sim = 1e9f;
    bool  detected = false;
    int   aborts = 0, restarts = 0;

    for (int step = 0; step < WALK_LEN; ++step) {
        // --- key schedule: key, k_samp, k_restart = split(key, 3) ---------
        // counts iota(6) -> halves [0,1,2],[3,4,5]; bits=[e0a,e1a,e2a,e0b,e1b,e2b]
        uint32_t e0a, e0b, e1a, e1b, e2a, e2b;
        tf2x32(key0, key1, 0u, 3u, e0a, e0b);
        tf2x32(key0, key1, 1u, 4u, e1a, e1b);
        tf2x32(key0, key1, 2u, 5u, e2a, e2b);
        const uint32_t ks0 = e2a, ks1 = e0b;         // k_samp
        const uint32_t kr0 = e1b, kr1 = e2b;         // k_restart
        key0 = e0a; key1 = e1a;                      // advanced key
        // samp_keys = split(k_samp, 3)
        uint32_t f0a, f0b, f1a, f1b, f2a, f2b;
        tf2x32(ks0, ks1, 0u, 3u, f0a, f0b);
        tf2x32(ks0, ks1, 1u, 4u, f1a, f1b);
        tf2x32(ks0, ks1, 2u, 5u, f2a, f2b);
        const uint32_t sk[3][2] = {{f0a, f1a}, {f2a, f0b}, {f1b, f2b}};

        // --- stage cur row in LDS (B-matrix source; avoids 16x redundancy) -
        const _Float16* curp = h16 + (size_t)cur * HIDDEN;
        {
            const v8h* src = (const v8h*)curp + lane * 4;
            v8h* dst = (v8h*)s_cur + lane * 4;
            dst[0] = src[0]; dst[1] = src[1]; dst[2] = src[2]; dst[3] = src[3];
        }
        // A-tile row pointers: window position p = tile*16 + (lane&15)
        const _Float16* rp0 = h16 + (size_t)clampi(cur - HALF_WIN + m,      0, SEQ_LEN - 1) * HIDDEN;
        const _Float16* rp1 = h16 + (size_t)clampi(cur - HALF_WIN + 16 + m, 0, SEQ_LEN - 1) * HIDDEN;
        const _Float16* rp2 = h16 + (size_t)clampi(cur - HALF_WIN + 32 + m, 0, SEQ_LEN - 1) * HIDDEN;
        __syncthreads();

        // --- sims via WMMA: D[16x16] = A[16x32] x B[32x16] over K=1024 -----
        // A fragment (ISA 7.12.2): lane<16 holds K {0..7,16..23}, lane>=16
// holds K {8..15,24..31} of row m. B: lanes 0-15 K {0..7,16..23} (col=lane),
// lanes 16-31 K {8..15,24..31} -- cur replicated in every column.
        v8f acc0 = {}, acc1 = {}, acc2 = {};
#pragma unroll 4
        for (int c = 0; c < 32; ++c) {
            const int kb = c * 32;
            const v8h blo = *(const v8h*)&s_cur[kb + hi * 8];
            const v8h bhi = *(const v8h*)&s_cur[kb + 16 + hi * 8];
            const v16h B = __builtin_shufflevector(blo, bhi,
                0, 1, 2, 3, 4, 5, 6, 7, 8, 9, 10, 11, 12, 13, 14, 15);

            const v8h a0l = *(const v8h*)(rp0 + kb + hi * 8);
            const v8h a0h = *(const v8h*)(rp0 + kb + 16 + hi * 8);
            const v16h A0 = __builtin_shufflevector(a0l, a0h,
                0, 1, 2, 3, 4, 5, 6, 7, 8, 9, 10, 11, 12, 13, 14, 15);
            acc0 = __builtin_amdgcn_wmma_f32_16x16x32_f16(
                false, A0, false, B, (short)0, acc0, false, false);

            const v8h a1l = *(const v8h*)(rp1 + kb + hi * 8);
            const v8h a1h = *(const v8h*)(rp1 + kb + 16 + hi * 8);
            const v16h A1 = __builtin_shufflevector(a1l, a1h,
                0, 1, 2, 3, 4, 5, 6, 7, 8, 9, 10, 11, 12, 13, 14, 15);
            acc1 = __builtin_amdgcn_wmma_f32_16x16x32_f16(
                false, A1, false, B, (short)0, acc1, false, false);

            const v8h a2l = *(const v8h*)(rp2 + kb + hi * 8);
            const v8h a2h = *(const v8h*)(rp2 + kb + 16 + hi * 8);
            const v16h A2 = __builtin_shufflevector(a2l, a2h,
                0, 1, 2, 3, 4, 5, 6, 7, 8, 9, 10, 11, 12, 13, 14, 15);
            acc2 = __builtin_amdgcn_wmma_f32_16x16x32_f16(
                false, A2, false, B, (short)0, acc2, false, false);
        }

        // C/D layout: lane 0 holds rows M=0..7 (VGPR r), lane 16 rows 8..15.
        if (lane == 0) {
#pragma unroll
            for (int r = 0; r < 8; ++r) {
                s_sims[r] = acc0[r]; s_sims[16 + r] = acc1[r]; s_sims[32 + r] = acc2[r];
            }
        } else if (lane == 16) {
#pragma unroll
            for (int r = 0; r < 8; ++r) {
                s_sims[8 + r] = acc0[r]; s_sims[24 + r] = acc1[r]; s_sims[40 + r] = acc2[r];
            }
        }
        __syncthreads();

        // --- 3 categorical attempts (Gumbel argmax over 33 logits) ---------
        // random_bits(key,(33,)): iota(33) padded -> lane j evals (j, 17+j)
        // (lane 16 pair (16,0)); y0 -> positions 0..16, y1 -> 17..32.
        int picks[3];
        for (int a = 0; a < 3; ++a) {
            if (lane < 17) {
                const uint32_t x1v = (lane < 16) ? (uint32_t)(17 + lane) : 0u;
                uint32_t g0, g1;
                tf2x32(sk[a][0], sk[a][1], (uint32_t)lane, x1v, g0, g1);
                s_score[lane] = logit_at(s_sims, cur, lane) + gumbel_from_bits(g0);
                if (lane < 16)
                    s_score[17 + lane] =
                        logit_at(s_sims, cur, 17 + lane) + gumbel_from_bits(g1);
            }
            __syncthreads();
            if (lane == 0) {                      // first-max == jnp.argmax
                int best = 0; float bv = s_score[0];
                for (int p = 1; p < WIN; ++p) {
                    const float sc = s_score[p];
                    if (sc > bv) { bv = sc; best = p; }
                }
                s_pick = best;
            }
            __syncthreads();
            picks[a] = s_pick;
        }

        // --- triadic reciprocity check + walk state update (uniform) -------
        const float rs_prev = rsq[prev];
        const float rs_cur  = rsq[cur];
        int cands[3]; bool ok[3];
#pragma unroll
        for (int a = 0; a < 3; ++a) {
            const int cnd = cur - HALF_WIN + picks[a];
            cands[a] = cnd;
            bool o = true;
            if (plen >= 2) {
                const float na = rs_prev, nb = rs_cur;
                const float nc = rsq[clampi(cnd, 0, SEQ_LEN - 1)];
                const float tr = (na * nb * nc) /
                                 ((na + EPS8) * (nb + EPS8) * (nc + EPS8));
                const float dev = fabsf(tr - rintf(tr));   // round-half-even
                o = (dev <= 0.1f) && (tr <= 1.5f);
            }
            ok[a] = o;
        }
        const bool found = ok[0] | ok[1] | ok[2];
        const int  first = ok[0] ? 0 : (ok[1] ? 1 : (ok[2] ? 2 : 0));
        const int  n_ab  = found ? first : 3;
        const int  cand  = cands[first];

        // randint(k_restart, (), 0, 1024): bits from eval (0,0); 2^32%1024==0
        uint32_t rb0, rb1;
        tf2x32(kr0, kr1, 0u, 0u, rb0, rb1);
        const int rnode = viol[rb0 & (NUM_VIOL - 1)];

        const int   pf  = (first == 0) ? picks[0] : ((first == 1) ? picks[1] : picks[2]);
        const float sim = s_sims[pf];
        const float new_min = fminf(min_sim, sim);
        const bool  closed  = found && (cand == start) && (plen > 2);

        aborts   += n_ab;
        restarts += found ? 0 : 1;
        if (closed && (new_min < 0.1f)) detected = true;

        __syncthreads();   // protect s_sims/s_cur before next-step rewrite
        if (closed) break; // 'done': remaining steps freeze all state

        if (found) { prev = cur; cur = cand; plen += 1; min_sim = new_min; }
        else       { cur = rnode; prev = rnode; plen = 1; }
    }

    if (lane == 0) {
        if (detected) out[v] = 1.0f;   // benign same-value race across 5 walks
        atomicAdd(out + NUM_VIOL,     (float)aborts);   // integer-valued: exact
        atomicAdd(out + NUM_VIOL + 1, (float)restarts);
    }
}

// ------------------------------- launcher ----------------------------------
extern "C" void kernel_launch(void* const* d_in, const int* in_sizes, int n_in,
                              void* d_out, int out_size, void* d_ws, size_t ws_size,
                              hipStream_t stream) {
    (void)in_sizes; (void)n_in; (void)ws_size;

    const float* hs  = (const float*)d_in[0];       // [8192,1024] fp32
    const int*   vio = (const int*)d_in[1];         // [1024] int32
    float*       out = (float*)d_out;               // 1024 det + aborts + restarts

    // workspace: fp16 normalized table (16 MB) + row_sq (32 KB)
    _Float16* h16 = (_Float16*)d_ws;
    float*    rsq = (float*)((char*)d_ws + (size_t)SEQ_LEN * HIDDEN * sizeof(_Float16));

    zero_out_kernel<<<(out_size + 255) / 256, 256, 0, stream>>>(out, out_size);
    prep_norm_kernel<<<SEQ_LEN, 256, 0, stream>>>(hs, h16, rsq);
    walk_kernel<<<NWALK_TOT, 32, 0, stream>>>(h16, rsq, vio, out);
}
